// LLaMAAttention_91182155694206
// MI455X (gfx1250) — compile-verified, matrix operand order checked by reading, not run on hardware
//
#include <hip/hip_runtime.h>
#include <hip/hip_bf16.h>

// LLaMA attention: B=32, T=256, C=2048, H=16, D=128.
#define B_ 32
#define T_ 256
#define C_ 2048
#define H_ 16
#define D_ 128

typedef __attribute__((ext_vector_type(16))) _Float16 v16h;
typedef __attribute__((ext_vector_type(8)))  _Float16 v8h;
typedef __attribute__((ext_vector_type(8)))  float    v8f;

// A/B fragment loader for v_wmma_f32_16x16x32_f16.
// 16-bit A (16x32) per-lane layout: lane L holds row M=L%16; K elements
// {kb..kb+7, kb+16..kb+23} where kb = (L/16)*8. B (32x16) is symmetric with
// N=L%16 columns. Both are two contiguous 16-byte chunks -> global_load_b128.
static __device__ inline v16h load_frag16(const _Float16* p, int kb) {
  v8h lo = *(const v8h*)(p + kb);
  v8h hi = *(const v8h*)(p + kb + 16);
  return __builtin_shufflevector(lo, hi, 0, 1, 2, 3, 4, 5, 6, 7,
                                 8, 9, 10, 11, 12, 13, 14, 15);
}

static __device__ inline v8f wmma16(v16h a, v16h b, v8f c) {
  return __builtin_amdgcn_wmma_f32_16x16x32_f16(
      /*neg_a=*/false, a, /*neg_b=*/false, b,
      /*c_mod=*/(short)0, c, /*reuse_a=*/false, /*reuse_b=*/false);
}

// ---------------------------------------------------------------- conversion
__global__ __launch_bounds__(256) void cvt_f32_f16(const float* __restrict__ s,
                                                   _Float16* __restrict__ d,
                                                   long n) {
  long i = (long)blockIdx.x * blockDim.x + threadIdx.x;
  long st = (long)gridDim.x * blockDim.x;
  for (; i < n; i += st) d[i] = (_Float16)s[i];
}

// ---------------------------------------------------------------- GEMM
// C[M,2048] = A[M,2048] * W[2048,2048]^T (both read contiguously along K).
// K and N are compile-time constants so all k-offsets fold into load
// immediates / uniform increments instead of per-lane 64-bit adds.
// Block = 256 thr = 8 waves (2 in M x 4 in N) -> 64x256 tile.
// Wave = 32x64 (2 M-tiles x 4 N-tiles): 8 WMMAs / 6 fragment loads per
// k-step (~21 FLOP/byte at cache level), software-pipelined with a
// double-buffered fragment set (prefetch k+32 while computing k).
// mode 0: store f16 [B,H,T,D]   (Q, K)
// mode 1: store f16 [B,H,D,T]   (V transposed for P*V B-fragments)
// mode 2: store f32 row-major [M,N]  (final projection to d_out)
__global__ __launch_bounds__(256) void gemm_wmma(const _Float16* __restrict__ A,
                                                 const _Float16* __restrict__ W,
                                                 void* __restrict__ dst,
                                                 int mode) {
  constexpr int K = C_;     // 2048
  constexpr int N = C_;     // 2048
  const int lane = threadIdx.x & 31;
  const int wave = threadIdx.x >> 5;
  const int wm = wave >> 2;          // 0..1
  const int wn = wave & 3;           // 0..3
  const int m0 = blockIdx.x * 64 + wm * 32;
  const int n0 = blockIdx.y * 256 + wn * 64;
  const int lrow = lane & 15;
  const int hi = lane >> 4;
  const int kb = hi << 3;

  v8f acc[2][4] = {};
  const _Float16* ap[2];
  const _Float16* bp[4];
#pragma unroll
  for (int ti = 0; ti < 2; ++ti) ap[ti] = A + (long)(m0 + ti * 16 + lrow) * K;
#pragma unroll
  for (int tj = 0; tj < 4; ++tj) bp[tj] = W + (long)(n0 + tj * 16 + lrow) * K;

  v16h aX[2], bX[4], aY[2], bY[4];   // double-buffered fragment sets

#define LOADSET(aa, bb, kk)                                            \
  do {                                                                 \
    _Pragma("unroll") for (int ti = 0; ti < 2; ++ti)                   \
        aa[ti] = load_frag16(ap[ti] + (kk), kb);                       \
    _Pragma("unroll") for (int tj = 0; tj < 4; ++tj)                   \
        bb[tj] = load_frag16(bp[tj] + (kk), kb);                       \
  } while (0)

#define MMASET(aa, bb)                                                 \
  do {                                                                 \
    _Pragma("unroll") for (int ti = 0; ti < 2; ++ti)                   \
        _Pragma("unroll") for (int tj = 0; tj < 4; ++tj)               \
            acc[ti][tj] = wmma16(aa[ti], bb[tj], acc[ti][tj]);         \
  } while (0)

  // K = 2048 = 64 k-steps: steady state processes k-steps in pairs so the
  // two buffers swap roles without register copies.
  LOADSET(aX, bX, 0);
  int k0 = 0;
  for (; k0 + 64 < K; k0 += 64) {
    LOADSET(aY, bY, k0 + 32);   // prefetch next k-step
    MMASET(aX, bX);             // compute current (loads already in flight)
    LOADSET(aX, bX, k0 + 64);
    MMASET(aY, bY);
  }
  LOADSET(aY, bY, k0 + 32);
  MMASET(aX, bX);
  MMASET(aY, bY);

#undef LOADSET
#undef MMASET

  // C/D layout: VGPR v, lanes 0-15 -> M=v, lanes 16-31 -> M=v+8; N=lane%16.
#pragma unroll
  for (int ti = 0; ti < 2; ++ti)
#pragma unroll
    for (int tj = 0; tj < 4; ++tj)
#pragma unroll
      for (int v = 0; v < 8; ++v) {
        int gm = m0 + ti * 16 + v + hi * 8;
        int gn = n0 + tj * 16 + lrow;
        float val = acc[ti][tj][v];
        if (mode == 2) {
          ((float*)dst)[(long)gm * N + gn] = val;
        } else {
          int bb = gm >> 8, t = gm & (T_ - 1);   // T_=256
          int h = gn >> 7, d = gn & (D_ - 1);    // D_=128
          long idx = (mode == 0)
                         ? (((long)(bb * H_ + h) * T_) + t) * D_ + d
                         : (((long)(bb * H_ + h) * D_) + d) * T_ + t;
          ((_Float16*)dst)[idx] = (_Float16)val;
        }
      }
}

// ---------------------------------------------------------------- RoPE
// Q,K in [B,H,T,D]; pair (d, d+64), angle = t * 10000^(-2d/128).
__global__ __launch_bounds__(256) void rope_kernel(_Float16* __restrict__ Q,
                                                   _Float16* __restrict__ Kp) {
  int i = blockIdx.x * blockDim.x + threadIdx.x;   // B*H*T*64 threads
  int p = i & 63;
  int t = (i >> 6) & (T_ - 1);
  int bh = i >> 14;
  long base = ((long)bh * T_ + t) * (long)D_;
  float inv = __expf(-(2.0f * (float)p / (float)D_) * 9.210340371976184f);
  float ang = (float)t * inv;
  float s, c;
  __sincosf(ang, &s, &c);
  float q0 = (float)Q[base + p], q1 = (float)Q[base + p + 64];
  Q[base + p]      = (_Float16)(q0 * c - q1 * s);
  Q[base + p + 64] = (_Float16)(q1 * c + q0 * s);
  float k0 = (float)Kp[base + p], k1 = (float)Kp[base + p + 64];
  Kp[base + p]      = (_Float16)(k0 * c - k1 * s);
  Kp[base + p + 64] = (_Float16)(k1 * c + k0 * s);
}

// ---------------------------------------------------------------- attention
// One wave per (b, h, 16-row q tile). Online (flash) softmax over kv tiles
// of 32 columns. S via 8 WMMAs (2 col-subtiles x 4 D-chunks); P goes through
// LDS to reshape into the A-fragment layout; P*V via 8 WMMAs into O[8] f32.
__global__ __launch_bounds__(128) void attn_kernel(const _Float16* __restrict__ Q,
                                                   const _Float16* __restrict__ Km,
                                                   const _Float16* __restrict__ Vt,
                                                   _Float16* __restrict__ Out) {
  __shared__ _Float16 Pl[4][16 * 32];   // 1 KB per wave
  const int lane = threadIdx.x & 31;
  const int wave = threadIdx.x >> 5;
  const int gw = blockIdx.x * 4 + wave;  // 0 .. B*H*16-1
  const int qt = gw & 15;                // q tile within head
  const int bh = gw >> 4;                // b*H + h
  const int b = bh >> 4;                 // H_=16
  const int h = bh & 15;
  const int lrow = lane & 15;
  const int hi = lane >> 4;
  const int kb = hi << 3;

  const _Float16* Qh = Q + (long)bh * T_ * D_;
  const _Float16* Kh = Km + (long)bh * T_ * D_;
  const _Float16* Vh = Vt + (long)bh * D_ * T_;

  v16h qf[4];
  {
    const _Float16* qr = Qh + (long)(qt * 16 + lrow) * D_;
#pragma unroll
    for (int c = 0; c < 4; ++c) qf[c] = load_frag16(qr + 32 * c, kb);
  }

  v8f O[8] = {};
  float mr[8], lr[8];
#pragma unroll
  for (int v = 0; v < 8; ++v) { mr[v] = -1e30f; lr[v] = 0.0f; }

  const int jn = (qt * 16 + 15) / 32 + 1;  // causal: only tiles touching k<=q
  for (int j = 0; j < jn; ++j) {
    v8f S[2] = {};
#pragma unroll
    for (int c = 0; c < 2; ++c) {
      const _Float16* kr = Kh + (long)(j * 32 + c * 16 + lrow) * D_;
#pragma unroll
      for (int kc = 0; kc < 4; ++kc) {
        v16h kf = load_frag16(kr + 32 * kc, kb);
        S[c] = wmma16(qf[kc], kf, S[c]);
      }
    }
    const float sc = 0.08838834764831845f;  // 1/sqrt(128)
#pragma unroll
    for (int v = 0; v < 8; ++v) {
      const int gq = qt * 16 + v + hi * 8;
      float s0 = S[0][v] * sc;
      float s1 = S[1][v] * sc;
      if (j * 32 + lrow > gq)      s0 = -1e30f;  // causal mask
      if (j * 32 + 16 + lrow > gq) s1 = -1e30f;
      float mx = fmaxf(s0, s1);
#pragma unroll
      for (int o = 1; o < 16; o <<= 1) mx = fmaxf(mx, __shfl_xor(mx, o, 16));
      const float mn = fmaxf(mr[v], mx);
      const float f = __expf(mr[v] - mn);
      mr[v] = mn;
      const float p0 = __expf(s0 - mn);
      const float p1 = __expf(s1 - mn);
      float rs = p0 + p1;
#pragma unroll
      for (int o = 1; o < 16; o <<= 1) rs += __shfl_xor(rs, o, 16);
      lr[v] = lr[v] * f + rs;
#pragma unroll
      for (int dt = 0; dt < 8; ++dt) O[dt][v] *= f;
      const int prow = v + hi * 8;
      Pl[wave][prow * 32 + lrow]      = (_Float16)p0;
      Pl[wave][prow * 32 + 16 + lrow] = (_Float16)p1;
    }
    asm volatile("s_wait_dscnt 0" ::: "memory");
    v16h pf;  // reload P in A-fragment layout (row = lane%16, chunks at kb)
    {
      const _Float16* pr = &Pl[wave][lrow * 32];
      v8h lo = *(const v8h*)(pr + kb);
      v8h hb = *(const v8h*)(pr + kb + 16);
      pf = __builtin_shufflevector(lo, hb, 0, 1, 2, 3, 4, 5, 6, 7,
                                   8, 9, 10, 11, 12, 13, 14, 15);
    }
#pragma unroll
    for (int dt = 0; dt < 8; ++dt) {
      const _Float16* vr = Vh + (long)(dt * 16 + lrow) * T_ + j * 32;
      v16h vf = load_frag16(vr, kb);
      O[dt] = wmma16(pf, vf, O[dt]);
    }
  }

  // normalize rows and store to [B, T, H, D] (== [B*T, C] for the final GEMM)
#pragma unroll
  for (int v = 0; v < 8; ++v) {
    const float inv = 1.0f / lr[v];
    const int row = qt * 16 + v + hi * 8;
    _Float16* orow = Out + ((long)(b * T_ + row)) * C_ + h * D_;
#pragma unroll
    for (int dt = 0; dt < 8; ++dt)
      orow[dt * 16 + lrow] = (_Float16)(O[dt][v] * inv);
  }
}

// ---------------------------------------------------------------- launch
extern "C" void kernel_launch(void* const* d_in, const int* in_sizes, int n_in,
                              void* d_out, int out_size, void* d_ws, size_t ws_size,
                              hipStream_t stream) {
  const float* x  = (const float*)d_in[0];
  const float* wq = (const float*)d_in[1];
  const float* wk = (const float*)d_in[2];
  const float* wv = (const float*)d_in[3];
  const float* wo = (const float*)d_in[4];
  float* out = (float*)d_out;

  const long MT  = (long)B_ * T_;   // 8192 tokens
  const long XSZ = MT * C_;         // 16,777,216 elements
  const long WSZ = (long)C_ * C_;   // 4,194,304 elements

  _Float16* ws  = (_Float16*)d_ws;
  _Float16* xb  = ws;  ws += XSZ;
  _Float16* wqb = ws;  ws += WSZ;
  _Float16* wkb = ws;  ws += WSZ;
  _Float16* wvb = ws;  ws += WSZ;
  _Float16* wob = ws;  ws += WSZ;
  _Float16* Qb  = ws;  ws += XSZ;
  _Float16* Kb  = ws;  ws += XSZ;
  _Float16* Vt  = ws;  ws += XSZ;
  _Float16* Ab  = ws;  ws += XSZ;   // attention output, [B*T, C] f16

  cvt_f32_f16<<<4096, 256, 0, stream>>>(x, xb, XSZ);
  cvt_f32_f16<<<2048, 256, 0, stream>>>(wq, wqb, WSZ);
  cvt_f32_f16<<<2048, 256, 0, stream>>>(wk, wkb, WSZ);
  cvt_f32_f16<<<2048, 256, 0, stream>>>(wv, wvb, WSZ);
  cvt_f32_f16<<<2048, 256, 0, stream>>>(wo, wob, WSZ);

  dim3 g((unsigned)(MT / 64), C_ / 256);   // 128 x 8 blocks
  gemm_wmma<<<g, 256, 0, stream>>>(xb, wqb, (void*)Qb, 0);
  gemm_wmma<<<g, 256, 0, stream>>>(xb, wkb, (void*)Kb, 0);
  gemm_wmma<<<g, 256, 0, stream>>>(xb, wvb, (void*)Vt, 1);

  rope_kernel<<<(B_ * H_ * T_ * 64) / 256, 256, 0, stream>>>(Qb, Kb);

  attn_kernel<<<(B_ * H_ * 16) / 4, 128, 0, stream>>>(Qb, Kb, Vt, Ab);

  gemm_wmma<<<g, 256, 0, stream>>>(Ab, wob, (void*)out, 2);
}
